// InvariantAttention_3874060501248
// MI455X (gfx1250) — matrix operand order repaired and checked by reading, not compiled, not run on hardware
//
#include <hip/hip_runtime.h>

#define HID 1024
#define HD 64
#define NH 16
#define SEQ 1024

typedef __attribute__((ext_vector_type(16))) _Float16 v16h;
typedef __attribute__((ext_vector_type(8)))  _Float16 v8h;
typedef __attribute__((ext_vector_type(8)))  float    v8f;

// ---- DPP16 butterfly reductions over each 16-lane half (wave32) ----------
template <int CTRL>
__device__ __forceinline__ float dpp_mov(float x) {
    int xi = __float_as_int(x);
    return __int_as_float(__builtin_amdgcn_update_dpp(xi, xi, CTRL, 0xF, 0xF, true));
}
__device__ __forceinline__ float red_max16(float v) {
    v = fmaxf(v, dpp_mov<0x140>(v));   // row_mirror      (pairs l <-> 15-l)
    v = fmaxf(v, dpp_mov<0x141>(v));   // row_half_mirror (pairs within 8)
    v = fmaxf(v, dpp_mov<0x1B>(v));    // quad_perm 3,2,1,0
    v = fmaxf(v, dpp_mov<0xB1>(v));    // quad_perm 1,0,3,2
    return v;
}
__device__ __forceinline__ float red_sum16(float v) {
    v += dpp_mov<0x140>(v);
    v += dpp_mov<0x141>(v);
    v += dpp_mov<0x1B>(v);
    v += dpp_mov<0xB1>(v);
    return v;
}

// ---------------------------------------------------------------------------
// Kernel 0: pack K -> f16 [b,h,s,d] and V -> f16 transposed [b,h,d,s]
// grid = 4096, 256 threads
// ---------------------------------------------------------------------------
__global__ __launch_bounds__(256) void pack_kv(
    const float* __restrict__ k, const float* __restrict__ v,
    _Float16* __restrict__ khf, _Float16* __restrict__ vtf)
{
    const int t = blockIdx.x * 256 + threadIdx.x;       // 1M threads
#pragma unroll
    for (int i = 0; i < 4; ++i) {                       // K: (b,s,h,d) order
        int e = t + i * (4096 * 256);
        int d = e & 63, h = (e >> 6) & 15, s = (e >> 10) & 1023, b = e >> 20;
        khf[((size_t)(b * NH + h) * SEQ + s) * HD + d] = (_Float16)k[e];
    }
#pragma unroll
    for (int i = 0; i < 4; ++i) {                       // V^T: (b,h,d,s) order
        int e = t + i * (4096 * 256);
        int s = e & 1023, d = (e >> 10) & 63, h = (e >> 16) & 15, b = e >> 20;
        vtf[e] = (_Float16)v[((size_t)(b * SEQ + s)) * HID + h * HD + d];
    }
}

// ---------------------------------------------------------------------------
// Kernel 1: fused attention, flash-style, async double-buffered K/V staging
// grid = B*NH*(T/128) = 512 blocks, 256 threads (8 waves, 16 query rows/wave)
// ---------------------------------------------------------------------------
__global__ __launch_bounds__(256) void attn_kernel(
    const float* __restrict__ q, const _Float16* __restrict__ khf,
    const _Float16* __restrict__ vtf, const float* __restrict__ bias,
    const unsigned char* __restrict__ mask, float* __restrict__ attn_out)
{
    __shared__ __align__(32) _Float16 Kf[2][32][64];   // K chunk (2 x 4 KB)
    __shared__ __align__(32) _Float16 Vt[2][64][32];   // V^T chunk (2 x 4 KB)
    __shared__ __align__(16) _Float16 Pl[8][16][32];   // per-wave P tile (8 KB)

    const int tid  = threadIdx.x;
    const int wid  = tid >> 5;
    const int lane = tid & 31;
    const int ln   = lane & 15;
    const int hl   = lane >> 4;

    const int blk = blockIdx.x;
    const int tb  = blk & 7;
    const int h   = (blk >> 3) & 15;
    const int b   = blk >> 7;
    const int bh  = b * NH + h;
    const int trow = tb * 128 + wid * 16;

    const _Float16* kbase = khf + (size_t)bh * (SEQ * HD);   // contiguous [s][d]
    const _Float16* vbase = vtf + (size_t)bh * (HD * SEQ);   // [d][s]

    // async stage one 32-key chunk into buffer `buf` (one B128 per thread x2)
    auto issue_chunk = [&](int sc, int buf) {
        unsigned kl = (unsigned)(size_t)&Kf[buf][0][0] + (unsigned)tid * 16u;
        unsigned long long ka =
            (unsigned long long)(size_t)(kbase + sc * (32 * 64)) + (size_t)tid * 16u;
        asm volatile("global_load_async_to_lds_b128 %0, %1, off"
                     :: "v"(kl), "v"(ka) : "memory");
        int d = tid >> 2, part = tid & 3;
        unsigned vl = (unsigned)(size_t)&Vt[buf][d][part * 8];
        unsigned long long va =
            (unsigned long long)(size_t)(vbase + (size_t)d * SEQ + sc * 32 + part * 8);
        asm volatile("global_load_async_to_lds_b128 %0, %1, off"
                     :: "v"(vl), "v"(va) : "memory");
    };

    // --- Q A-fragments: 16 rows x 64 dims, scaled by 1/8, f16 --------------
    v16h Aq[2];
    {
        const float* qrow = q + (size_t)(b * SEQ + trow + ln) * HID + h * HD;
#pragma unroll
        for (int c = 0; c < 2; ++c) {
#pragma unroll
            for (int e = 0; e < 8; ++e) {
                Aq[c][e]     = (_Float16)(qrow[c * 32 + hl * 16 + e]      * 0.125f);
                Aq[c][e + 8] = (_Float16)(qrow[c * 32 + hl * 16 + 8 + e] * 0.125f);
            }
        }
    }

    v8f O[4];
#pragma unroll
    for (int d = 0; d < 4; ++d) O[d] = (v8f){0,0,0,0,0,0,0,0};
    float mrow[8], lrow[8];
#pragma unroll
    for (int r = 0; r < 8; ++r) { mrow[r] = -INFINITY; lrow[r] = 0.0f; }

    const float* bbase = bias + ((size_t)(bh * SEQ + trow)) * SEQ;

    issue_chunk(0, 0);

    for (int sc = 0; sc < 32; ++sc) {
        const int s0  = sc * 32;
        const int cur = sc & 1;
        if (sc + 1 < 32) {
            issue_chunk(sc + 1, cur ^ 1);
            asm volatile("s_wait_asynccnt 2" ::: "memory");  // chunk sc landed
        } else {
            asm volatile("s_wait_asynccnt 0" ::: "memory");
        }
        __syncthreads();

        // --- scores: two 16x16 N-tiles, K=64 -> 2 WMMAs each ---------------
        v8f Sj[2];
#pragma unroll
        for (int j = 0; j < 2; ++j) {
            v8f acc = (v8f){0,0,0,0,0,0,0,0};
            v16h B0 = *(const v16h*)&Kf[cur][j * 16 + ln][hl * 16];
            v16h B1 = *(const v16h*)&Kf[cur][j * 16 + ln][32 + hl * 16];
            acc = __builtin_amdgcn_wmma_f32_16x16x32_f16(false, Aq[0], false, B0,
                                                         (short)0, acc, false, false);
            acc = __builtin_amdgcn_wmma_f32_16x16x32_f16(false, Aq[1], false, B1,
                                                         (short)0, acc, false, false);
            Sj[j] = acc;
        }

        // --- add bias (NT stream) + key padding mask -----------------------
        const unsigned char mk[2] = { mask[b * SEQ + s0 + ln],
                                      mask[b * SEQ + s0 + 16 + ln] };
#pragma unroll
        for (int j = 0; j < 2; ++j) {
            int col = s0 + j * 16 + ln;
#pragma unroll
            for (int r = 0; r < 8; ++r) {
                int M = r + hl * 8;
                float sv = Sj[j][r] +
                    __builtin_nontemporal_load(&bbase[(size_t)M * SEQ + col]);
                Sj[j][r] = mk[j] ? -INFINITY : sv;
            }
        }

        // --- online softmax (DPP butterfly reductions, no LDS) -------------
        float alpha[8];
#pragma unroll
        for (int r = 0; r < 8; ++r) {
            float cm = red_max16(fmaxf(Sj[0][r], Sj[1][r]));
            float mn = fmaxf(mrow[r], cm);
            float a  = __expf(mrow[r] - mn);
            float p0 = __expf(Sj[0][r] - mn);
            float p1 = __expf(Sj[1][r] - mn);
            float rs = red_sum16(p0 + p1);
            lrow[r] = lrow[r] * a + rs;
            mrow[r] = mn;
            alpha[r] = a;
            Sj[0][r] = p0; Sj[1][r] = p1;
        }
#pragma unroll
        for (int d = 0; d < 4; ++d)
#pragma unroll
            for (int r = 0; r < 8; ++r) O[d][r] *= alpha[r];

        // --- P: C-frag layout -> A-frag layout via per-wave LDS ------------
#pragma unroll
        for (int j = 0; j < 2; ++j)
#pragma unroll
            for (int r = 0; r < 8; ++r)
                Pl[wid][r + hl * 8][j * 16 + ln] = (_Float16)Sj[j][r];
        asm volatile("s_wait_dscnt 0" ::: "memory");
        v16h Ap;
        {
            const _Float16* prow = &Pl[wid][ln][0];
            v8h lo = *(const v8h*)&prow[hl * 16];
            v8h hi = *(const v8h*)&prow[hl * 16 + 8];
#pragma unroll
            for (int e = 0; e < 8; ++e) { Ap[e] = lo[e]; Ap[e + 8] = hi[e]; }
        }

        // --- O += P @ V : 4 N-tiles over head dim --------------------------
#pragma unroll
        for (int d = 0; d < 4; ++d) {
            v16h Bv = *(const v16h*)&Vt[cur][d * 16 + ln][hl * 16];
            O[d] = __builtin_amdgcn_wmma_f32_16x16x32_f16(false, Ap, false, Bv,
                                                          (short)0, O[d], false, false);
        }
        __syncthreads();   // all waves done with buffer `cur` before refill
    }

    // --- normalize by row sum, write pre-LN attn (fp32) to workspace -------
#pragma unroll
    for (int r = 0; r < 8; ++r) {
        float inv = 1.0f / lrow[r];
        int t = trow + r + hl * 8;
#pragma unroll
        for (int d = 0; d < 4; ++d)
            attn_out[(size_t)(b * SEQ + t) * HID + h * HD + d * 16 + ln] =
                O[d][r] * inv;
    }
}

// ---------------------------------------------------------------------------
// Kernel 2: LayerNorm per row (one wave per row), fp32 -> f16 normalized acts
// grid = 512, 256 threads
// ---------------------------------------------------------------------------
__global__ __launch_bounds__(256) void ln_kernel(
    const float* __restrict__ x, const float* __restrict__ lw,
    const float* __restrict__ lb, _Float16* __restrict__ xn)
{
    const int row  = blockIdx.x * 8 + (threadIdx.x >> 5);
    const int lane = threadIdx.x & 31;
    const float* xr = x + (size_t)row * HID;
    float s = 0.0f, sq = 0.0f;
    float vals[32];
#pragma unroll
    for (int i = 0; i < 32; ++i) {
        float t = xr[lane + i * 32];
        vals[i] = t; s += t; sq += t * t;
    }
#pragma unroll
    for (int off = 16; off >= 1; off >>= 1) {
        s  += __shfl_xor(s,  off, 32);
        sq += __shfl_xor(sq, off, 32);
    }
    float mu   = s * (1.0f / 1024.0f);
    float var  = sq * (1.0f / 1024.0f) - mu * mu;
    float rstd = rsqrtf(var + 1e-5f);
#pragma unroll
    for (int i = 0; i < 32; ++i) {
        int c = lane + i * 32;
        xn[(size_t)row * HID + c] =
            (_Float16)((vals[i] - mu) * rstd * lw[c] + lb[c]);
    }
}

// ---------------------------------------------------------------------------
// Kernel 3: out = xn @ W^T + b  (WMMA GEMM, 128x128 tiles)
// grid = (32, 8), 256 threads
// ---------------------------------------------------------------------------
__global__ __launch_bounds__(256) void proj_kernel(
    const _Float16* __restrict__ xn, const float* __restrict__ W,
    const float* __restrict__ ob, float* __restrict__ out)
{
    __shared__ __align__(32) _Float16 Wt[128][32];   // W tile, f16 (8 KB)
    const int tid  = threadIdx.x;
    const int wid  = tid >> 5;
    const int lane = tid & 31;
    const int ln   = lane & 15;
    const int hl   = lane >> 4;
    const int m0   = blockIdx.x * 128;
    const int o0   = blockIdx.y * 128;
    const int trow = m0 + wid * 16;

    v8f acc[8];
#pragma unroll
    for (int d = 0; d < 8; ++d) acc[d] = (v8f){0,0,0,0,0,0,0,0};

    for (int kc = 0; kc < 32; ++kc) {
        __syncthreads();
#pragma unroll
        for (int it = 0; it < 16; ++it) {
            int e = tid + it * 256;
            int ol = e >> 5, kl = e & 31;
            Wt[ol][kl] = (_Float16)W[(size_t)(o0 + ol) * HID + kc * 32 + kl];
        }
        __syncthreads();

        v16h A;
        {
            const _Float16* xr = xn + (size_t)(trow + ln) * HID + kc * 32;
            v8h lo = *(const v8h*)&xr[hl * 16];
            v8h hi = *(const v8h*)&xr[hl * 16 + 8];
#pragma unroll
            for (int e = 0; e < 8; ++e) { A[e] = lo[e]; A[e + 8] = hi[e]; }
        }
#pragma unroll
        for (int d = 0; d < 8; ++d) {
            v16h B = *(const v16h*)&Wt[d * 16 + ln][hl * 16];
            acc[d] = __builtin_amdgcn_wmma_f32_16x16x32_f16(false, A, false, B,
                                                            (short)0, acc[d], false, false);
        }
    }

#pragma unroll
    for (int d = 0; d < 8; ++d) {
        int o = o0 + d * 16 + ln;
        float bo = ob[o];
#pragma unroll
        for (int r = 0; r < 8; ++r) {
            int t = trow + r + hl * 8;
            out[(size_t)t * HID + o] = acc[d][r] + bo;
        }
    }
}

// ---------------------------------------------------------------------------
extern "C" void kernel_launch(void* const* d_in, const int* in_sizes, int n_in,
                              void* d_out, int out_size, void* d_ws, size_t ws_size,
                              hipStream_t stream)
{
    const float*         q    = (const float*)d_in[0];
    const float*         k    = (const float*)d_in[1];
    const float*         v    = (const float*)d_in[2];
    const float*         bias = (const float*)d_in[3];
    const unsigned char* mask = (const unsigned char*)d_in[4];
    const float*         outw = (const float*)d_in[5];
    const float*         outb = (const float*)d_in[6];
    const float*         lnw  = (const float*)d_in[7];
    const float*         lnb  = (const float*)d_in[8];
    float* out = (float*)d_out;

    float*    attn_ws = (float*)d_ws;                                   // 16 MB
    _Float16* xn_ws   = (_Float16*)((char*)d_ws + ((size_t)16 << 20));  //  8 MB
    _Float16* khf     = (_Float16*)((char*)d_ws + ((size_t)24 << 20));  //  8 MB
    _Float16* vtf     = (_Float16*)((char*)d_ws + ((size_t)32 << 20));  //  8 MB

    pack_kv   <<<4096, 256, 0, stream>>>(k, v, khf, vtf);
    attn_kernel<<<512, 256, 0, stream>>>(q, khf, vtf, bias, mask, attn_ws);
    ln_kernel  <<<512, 256, 0, stream>>>(attn_ws, lnw, lnb, xn_ws);
    proj_kernel<<<dim3(32, 8), 256, 0, stream>>>(xn_ws, outw, outb, out);
}
